// xLSTM_85641647882831
// MI455X (gfx1250) — matrix operand order repaired and checked by reading, not compiled, not run on hardware
//
#include <hip/hip_runtime.h>
#include <hip/hip_bf16.h>

typedef __attribute__((ext_vector_type(16))) _Float16 v16h;
typedef __attribute__((ext_vector_type(8)))  _Float16 v8h;
typedef __attribute__((ext_vector_type(8)))  float    v8f;

#define B_  16
#define T_  256
#define IN_ 19
#define H_  256

// ---------------- d_ws layout (float units) --------------------------------
#define MG_OFF   0                       // [T][B][6][H] fp32   : 6,291,456
#define HSF_OFF  6291456                 // [B][H] fp32         : 4,096
#define BAR_OFF  6295552                 // 256 uint counters   : 256
#define WCAT_OFF 6295808                 // f16 [1024][512]     : 262,144 floats
#define HBUF_OFF 6557952                 // f16 [2][2][16][256] : 8,192 floats

// ---------------------------------------------------------------------------
// Zero the per-step cross-workgroup arrival counters (graph-capture safe).
// ---------------------------------------------------------------------------
__global__ void zero_bar(unsigned* __restrict__ bar) { bar[threadIdx.x] = 0u; }

// ---------------------------------------------------------------------------
// Kernel A: mLSTM gate pre-activations for ALL timesteps (state independent).
//   mg[t][b][g][o] = sum_k x[b][t][k] * mW[g][k][o] + mb[g][o]
// M = T*B = 4096 (row r = t*16+b), K = 19 -> pad 32, N = 6H = 1536.
// One v_wmma_f32_16x16x32_f16 per 16x16 tile; each M-tile == one timestep.
// ---------------------------------------------------------------------------
__global__ __launch_bounds__(256)
void mgate_wmma(const float* __restrict__ x, const float* __restrict__ mW,
                const float* __restrict__ mb, float* __restrict__ mg) {
  const int lane = threadIdx.x & 31;
  const int wave = blockIdx.x * 8 + (threadIdx.x >> 5);   // 24576 waves
  const int tile_m = wave / 96;          // == t
  const int tile_n = wave % 96;
  const int row = lane & 15;
  const int grp = lane >> 4;

  const int t = tile_m;
  const int g = tile_n >> 4;
  const int o = ((tile_n & 15) << 4) + row;

  v16h a;
#pragma unroll
  for (int j = 0; j < 16; ++j) {      // A 16x32 f16 layout (K-interleave)
    int kk = (j & 7) + (grp << 3) + ((j >> 3) << 4);
    float av = (kk < IN_) ? x[(row * T_ + t) * IN_ + kk] : 0.0f;
    a[j] = (_Float16)av;
  }
  v16h bf;
#pragma unroll
  for (int j = 0; j < 16; ++j) {      // B 32x16 f16 layout (K-interleave, matches A)
    int kk = (j & 7) + (grp << 3) + ((j >> 3) << 4);
    float bv = (kk < IN_) ? mW[(g * IN_ + kk) * H_ + o] : 0.0f;
    bf[j] = (_Float16)bv;
  }
  v8f c = {};
  c = __builtin_amdgcn_wmma_f32_16x16x32_f16(false, a, false, bf,
                                             (short)0, c, false, false);
  const float bias = mb[g * H_ + o];
#pragma unroll
  for (int v = 0; v < 8; ++v) {
    int bb = v + (grp << 3);          // D: M = v + 8*grp
    mg[(((size_t)(t * B_ + bb) * 6 + g) * H_) + o] = c[v] + bias;
  }
}

// ---------------------------------------------------------------------------
// One-time conversion of sLSTM weights into WMMA-friendly f16 layout:
//   wcatT[n][k], n = g*256+oo (output col), k = 0..511 (concat [sW;sR] K dim),
// K-contiguous so each 16-half B-fragment run is one 32B contiguous load.
// ---------------------------------------------------------------------------
__global__ __launch_bounds__(512)
void prep_wcat(const float* __restrict__ sW, const float* __restrict__ sR,
               _Float16* __restrict__ wcatT) {
  int idx = blockIdx.x * 512 + threadIdx.x;       // 1024*512 elements
  int n = idx >> 9, k = idx & 511;
  int g = n >> 8, oo = n & 255;
  float v = (k < 256) ? sW[((size_t)g * 256 + k) * 256 + oo]
                      : sR[((size_t)g * 256 + (k - 256)) * 256 + oo];
  wcatT[idx] = (_Float16)v;
}

// ---------------------------------------------------------------------------
// Kernel B: the sequential recurrence. 16 workgroups, each 314KB LDS => one
// per WGP, 16 co-resident WGPs (safe on MI455X).
//   * mLSTM phase: batch-partitioned; the 256KB matrix memory C lives
//     entirely in CDNA5's 320KB LDS (row-skewed, bank-conflict free).
//   * sLSTM phase: column-partitioned batched GEMM [16,512]x[512,1024] done
//     with v_wmma_f32_16x16x32_f16 (4 waves x 4 K-chunks x 4 gate tiles).
//   * Cross-WG exchange via double-buffered f16 H panels + per-step global
//     arrival counter (release/acquire atomics, s_sleep spin).
// ---------------------------------------------------------------------------
__global__ __launch_bounds__(512)
void recurrence(const float* __restrict__ mg,
                const _Float16* __restrict__ wcatT,
                const float* __restrict__ sb,
                _Float16* __restrict__ hbuf,      // [2][2][16][256] f16
                unsigned* __restrict__ bar,       // [256] arrival counters
                float* __restrict__ hsfinal) {
  extern __shared__ float sm[];
  // ---- LDS map (floats) ---------------------------------------------------
  float*    C    = sm;                    // 65536 : C[i][(j+i)&255]
  float*    Q    = sm + 65536;            // 9 x 256 scalars
  float*    KK   = Q    + 256;
  float*    VV   = KK   + 256;
  float*    FP   = VV   + 256;
  float*    IP   = FP   + 256;
  float*    OT   = IP   + 256;
  float*    MST  = OT   + 256;
  float*    NST  = MST  + 256;
  float*    HM   = NST  + 256;
  float*    RED  = HM   + 256;            // 256
  float*    NUMP = RED  + 256;            // 512
  float*    DEN  = NUMP + 512;            // 16
  _Float16* Apan = (_Float16*)(DEN + 16); // 8192 halves (4096 floats), skewed
  float*    GP   = (float*)(Apan + 8192); // 4096 : per-K-chunk partial tiles
  float*    GT   = GP + 4096;             // 1024 : combined gates
  float*    CS2  = GT + 1024;             // sLSTM state slice [b*16+ol]
  float*    NS2  = CS2 + 256;
  float*    MS2  = NS2 + 256;             // total 78,608 floats = 314,432 B

  const int tid  = threadIdx.x;
  const int wg   = blockIdx.x;            // batch b for mLSTM, col-slice for sLSTM
  const int wave = tid >> 5;
  const int lane = tid & 31;
  const int row  = lane & 15;
  const int grp  = lane >> 4;

  for (int idx = tid; idx < 65536; idx += 512) C[idx] = 0.0f;
  if (tid < 256) {
    MST[tid] = 0.0f; NST[tid] = 0.0f; HM[tid] = 0.0f;
    CS2[tid] = 0.0f; NS2[tid] = 0.0f; MS2[tid] = 0.0f;
    hbuf[4096 + wg * 256 + tid] = (_Float16)0.0f;   // HS(-1) = 0 in buffer 0
  }
  __syncthreads();

  for (int t = 0; t < T_; ++t) {
    const float* gb = mg + (size_t)(t * B_ + wg) * 6 * H_;
    _Float16* hb_cur = hbuf + (size_t)(t & 1) * 8192;       // read panel
    _Float16* hb_nxt = hbuf + (size_t)((t + 1) & 1) * 8192; // HS(t) dest

    // (a) mLSTM gating + log-space stabilizer for own batch row
    if (tid < 256) {
      int i = tid;
      float q  = gb[i];
      float k  = gb[H_ + i] * 0.0625f;               // 1/sqrt(256)
      float v  = gb[2 * H_ + i];
      float it = gb[3 * H_ + i];
      float ft = gb[4 * H_ + i];
      float ot = 1.0f / (1.0f + __expf(-gb[5 * H_ + i]));
      float mo = MST[i];
      float mn = fmaxf(ft + mo, it);
      float ip = __expf(it - mn);
      float fp = __expf(ft + mo - mn);
      MST[i] = mn;
      float nn = fmaf(fp, NST[i], ip * k);
      NST[i] = nn;
      Q[i] = q; KK[i] = k; VV[i] = v; FP[i] = fp; IP[i] = ip; OT[i] = ot;
      RED[i] = nn * q;
    }
    __syncthreads();

    // (b) fused C update + num = C_new @ q (skewed, conflict-free)
    {
      int i  = tid & 255;
      int j0 = (tid >> 8) << 7;
      float fpi = FP[i];
      float ipv = IP[i] * VV[i];
      float acc = 0.0f;
#pragma unroll 4
      for (int jj = 0; jj < 128; ++jj) {
        int j    = j0 + jj;
        int addr = (i << 8) + ((j + i) & 255);
        float cn = fmaf(fpi, C[addr], ipv * KK[j]);
        C[addr]  = cn;
        acc = fmaf(cn, Q[j], acc);
      }
      NUMP[tid] = acc;
    }
    __syncthreads();

    if (tid < 32) {                       // den reduction (GT is free here)
      float s = 0.0f;
#pragma unroll
      for (int u = 0; u < 8; ++u) s += RED[tid * 8 + u];
      GT[tid] = s;
    }
    __syncthreads();
    if (tid == 0) {
      float s = 0.0f;
#pragma unroll
      for (int u = 0; u < 32; ++u) s += GT[u];
      DEN[0] = fmaxf(fabsf(s), 1.0f);
    }
    __syncthreads();

    // publish HM(t) as f16 into the current exchange panel
    if (tid < 256) {
      float h = OT[tid] * (NUMP[tid] + NUMP[tid + 256]) / DEN[0];
      HM[tid] = h;
      hb_cur[wg * 256 + tid] = (_Float16)h;
    }

    // ---- cross-workgroup barrier: all HM(t) / HS(t-1) visible -------------
    __threadfence();
    __syncthreads();
    if (tid == 0) {
      __hip_atomic_fetch_add(bar + t, 1u, __ATOMIC_ACQ_REL,
                             __HIP_MEMORY_SCOPE_AGENT);
      while (__hip_atomic_load(bar + t, __ATOMIC_ACQUIRE,
                               __HIP_MEMORY_SCOPE_AGENT) < (unsigned)B_)
        __builtin_amdgcn_s_sleep(1);
    }
    __syncthreads();

    // stage A panel [16 x 512] f16 into LDS, skewed by 8 halves per row so
    // fragment ds_load_b128s hit all 64 banks
#pragma unroll
    for (int rep = 0; rep < 2; ++rep) {
      int ch = tid + rep * 512;           // 1024 chunks of 8 halves
      int bb = ch >> 6;
      int k0 = (ch & 63) << 3;
      const _Float16* src = (k0 < 256) ? (hb_cur + bb * 256 + k0)
                                       : (hb_cur + 4096 + bb * 256 + (k0 - 256));
      *(v8h*)(Apan + bb * 512 + ((k0 + 8 * bb) & 511)) = *(const v8h*)src;
    }
    __syncthreads();

    // ---- batched sLSTM GEMM: [16,512] x [512, 64-col slice] ---------------
    if (wave < 4) {
      const int kw = wave;                // K-chunk 128 each
      v8f acc0 = {}, acc1 = {}, acc2 = {}, acc3 = {};
#pragma unroll
      for (int kf = 0; kf < 4; ++kf) {
        const int kbase = kw * 128 + kf * 32;
        // A fragment: row = batch, runs of 8 halves at +16*grp and +8+16*grp
        v8h alo = *(const v8h*)(Apan + row * 512 + ((kbase +     16 * grp + 8 * row) & 511));
        v8h ahi = *(const v8h*)(Apan + row * 512 + ((kbase + 8 + 16 * grp + 8 * row) & 511));
        v16h a = __builtin_shufflevector(alo, ahi, 0, 1, 2, 3, 4, 5, 6, 7,
                                         8, 9, 10, 11, 12, 13, 14, 15);
        const _Float16* wb = wcatT + ((size_t)(wg * 16 + row)) * 512 + kbase + 16 * grp;
#pragma unroll
        for (int g = 0; g < 4; ++g) {
          v16h bfrag = *(const v16h*)(wb + (size_t)g * 256 * 512);
          v8f* acc = (g == 0) ? &acc0 : (g == 1) ? &acc1 : (g == 2) ? &acc2 : &acc3;
          *acc = __builtin_amdgcn_wmma_f32_16x16x32_f16(false, a, false, bfrag,
                                                        (short)0, *acc, false, false);
        }
      }
#pragma unroll
      for (int v = 0; v < 8; ++v) {       // D: (b = v+8*grp, ol = row)
        int pos = (v + 8 * grp) * 16 + row;
        GP[kw * 1024 +       pos] = acc0[v];
        GP[kw * 1024 + 256 + pos] = acc1[v];
        GP[kw * 1024 + 512 + pos] = acc2[v];
        GP[kw * 1024 + 768 + pos] = acc3[v];
      }
    } else if (wave >= 8 && t + 1 < T_) {
      // idle waves prefetch next step's mLSTM gates (global_prefetch_b8)
      const float* nxt = mg + (size_t)((t + 1) * B_ + wg) * 6 * H_;
      __builtin_prefetch(nxt + (tid - 256) * 6, 0, 0);
    }
    __syncthreads();

    // combine 4 K-chunk partials + bias
#pragma unroll
    for (int rep = 0; rep < 2; ++rep) {
      int n = tid + rep * 512;            // 0..1023 = g*256 + b*16 + ol
      float s = GP[n] + GP[1024 + n] + GP[2048 + n] + GP[3072 + n];
      GT[n] = s + sb[(n >> 8) * H_ + wg * 16 + (n & 15)];
    }
    __syncthreads();

    // sLSTM pointwise update for owned (b, column) slice
    if (tid < 256) {
      int pos = tid;                      // b*16 + ol
      int bb = pos >> 4, ol = pos & 15;
      float z  = tanhf(GT[pos]);
      float it = GT[256 + pos];
      float ft = GT[512 + pos];
      float ot = 1.0f / (1.0f + __expf(-GT[768 + pos]));
      float mo = MS2[pos];
      float mn = fmaxf(ft + mo, it);
      float ip = __expf(it - mn);
      float fp = __expf(ft + mo - mn);
      float cn = fmaf(fp, CS2[pos], ip * z);
      float nn = fmaf(fp, NS2[pos], ip);
      MS2[pos] = mn; CS2[pos] = cn; NS2[pos] = nn;
      float h = ot * cn / nn;
      hb_nxt[4096 + bb * 256 + wg * 16 + ol] = (_Float16)h;   // HS(t)
      if (t == T_ - 1) hsfinal[bb * 256 + wg * 16 + ol] = h;
    }
    __syncthreads();
  }
}

// ---------------------------------------------------------------------------
// Kernel C: MLP head 256 -> 128 -> 64 -> 1 over B=16 rows. Tiny; one block.
// ---------------------------------------------------------------------------
__global__ __launch_bounds__(256)
void mlp_head(const float* __restrict__ hsf,
              const float* __restrict__ f1w, const float* __restrict__ f1b,
              const float* __restrict__ f2w, const float* __restrict__ f2b,
              const float* __restrict__ f3w, const float* __restrict__ f3b,
              float* __restrict__ out) {
  __shared__ float t1[16 * 128];
  __shared__ float t2[16 * 64];
  const int tid = threadIdx.x;

  for (int idx = tid; idx < 16 * 128; idx += 256) {
    int b = idx >> 7, o = idx & 127;
    float acc = f1b[o];
    for (int i = 0; i < 256; ++i) acc = fmaf(hsf[b * 256 + i], f1w[i * 128 + o], acc);
    t1[idx] = fmaxf(acc, 0.0f);
  }
  __syncthreads();
  for (int idx = tid; idx < 16 * 64; idx += 256) {
    int b = idx >> 6, o = idx & 63;
    float acc = f2b[o];
    for (int i = 0; i < 128; ++i) acc = fmaf(t1[b * 128 + i], f2w[i * 64 + o], acc);
    t2[idx] = fmaxf(acc, 0.0f);
  }
  __syncthreads();
  if (tid < 16) {
    float acc = f3b[0];
    for (int i = 0; i < 64; ++i) acc = fmaf(t2[tid * 64 + i], f3w[i], acc);
    out[tid] = acc;
  }
}

extern "C" void kernel_launch(void* const* d_in, const int* in_sizes, int n_in,
                              void* d_out, int out_size, void* d_ws, size_t ws_size,
                              hipStream_t stream) {
  (void)in_sizes; (void)n_in; (void)out_size; (void)ws_size;
  const float* x   = (const float*)d_in[0];
  const float* mW  = (const float*)d_in[1];
  const float* mb  = (const float*)d_in[2];
  const float* sW  = (const float*)d_in[3];
  const float* sR  = (const float*)d_in[4];
  const float* sb  = (const float*)d_in[5];
  const float* f1w = (const float*)d_in[6];
  const float* f1b = (const float*)d_in[7];
  const float* f2w = (const float*)d_in[8];
  const float* f2b = (const float*)d_in[9];
  const float* f3w = (const float*)d_in[10];
  const float* f3b = (const float*)d_in[11];

  float*     ws    = (float*)d_ws;
  float*     mg    = ws + MG_OFF;
  float*     hsf   = ws + HSF_OFF;
  unsigned*  bar   = (unsigned*)(ws + BAR_OFF);
  _Float16*  wcatT = (_Float16*)(ws + WCAT_OFF);
  _Float16*  hbuf  = (_Float16*)(ws + HBUF_OFF);

  zero_bar<<<1, 256, 0, stream>>>(bar);
  mgate_wmma<<<3072, 256, 0, stream>>>(x, mW, mb, mg);
  prep_wcat<<<1024, 512, 0, stream>>>(sW, sR, wcatT);

  const size_t lds_bytes = 78608u * sizeof(float);   // 314,432 B < 320KB WGP LDS
  recurrence<<<B_, 512, lds_bytes, stream>>>(mg, wcatT, sb, hbuf, bar, hsf);

  mlp_head<<<1, 256, 0, stream>>>(hsf, f1w, f1b, f2w, f2b, f3w, f3b, (float*)d_out);
}
